// CommNet_60490319397058
// MI455X (gfx1250) — matrix operand order repaired and checked by reading, not compiled, not run on hardware
//
#include <hip/hip_runtime.h>
#include <hip/hip_bf16.h>

// ---------------------------------------------------------------------------
// CommNet forward for MI455X (gfx1250, wave32, WMMA).
//   h = tanh(x @ W_enc + b_enc)
//   repeat K=2: c = mean_agents(h); h = tanh(h @ W_top + (c @ W_bot + b_h)[batch])
//   out = h @ W_dec + b_dec
// All GEMMs use v_wmma_f32_16x16x32_f16 (f16 in, f32 accumulate).
// Each wave32 owns a 32x64 output tile (MT=2 x NT=4). The weight matrix BT is
// staged once per workgroup into LDS and B fragments are read via ds_load_b128;
// A streams from global (per-block A stripe fits L0). K loop is unrolled by 2
// only, to bound the software-pipeline window and avoid scratch spills.
// ---------------------------------------------------------------------------

typedef __attribute__((ext_vector_type(8)))  _Float16 v8h;
typedef __attribute__((ext_vector_type(16))) _Float16 v16h;
typedef __attribute__((ext_vector_type(8)))  float    v8f;

#define B_SZ   4096
#define A_AG   50
#define DIN    128
#define HDIM   256
#define DOUT   64
#define KCOMM  2
#define MROWS  (B_SZ * A_AG)   // 204800
#define MT     2               // 16-row tiles per wave  -> 32 rows
#define NT     4               // 16-col tiles per wave  -> 64 cols

// Branch-free tanh: 1 - 2/(e^{2x}+1). v_exp_f32 + v_rcp_f32, no EXEC games.
// Saturates exactly to +/-1 for large |x| (rcp(inf)=0, rcp(1)=1).
__device__ __forceinline__ float fast_tanh(float x) {
    float e = __expf(2.0f * x);
    return 1.0f - 2.0f * __builtin_amdgcn_rcpf(e + 1.0f);
}

// ---------------------------------------------------------------------------
// Elementwise fp32 -> f16 conversion.
__global__ void cvt_f32_to_f16(const float* __restrict__ src,
                               _Float16* __restrict__ dst, long n) {
    long i = (long)blockIdx.x * blockDim.x + threadIdx.x;
    if (i < n) dst[i] = (_Float16)src[i];
}

// Transpose + convert: dst[c*R + r] = (f16) src[r*C + c].  (weights are tiny)
__global__ void transpose_f32_to_f16(const float* __restrict__ src,
                                     _Float16* __restrict__ dst, int R, int C) {
    int i = blockIdx.x * blockDim.x + threadIdx.x;
    if (i < R * C) {
        int r = i / C, c = i % C;
        dst[(size_t)c * R + r] = (_Float16)src[(size_t)r * C + c];
    }
}

// Per-batch mean over agents: c[b][j] = mean_a h[(b*A + a)][j], f16 in/out.
__global__ void mean_agents(const _Float16* __restrict__ h,
                            _Float16* __restrict__ c) {
    int i = blockIdx.x * blockDim.x + threadIdx.x;  // over B_SZ * HDIM
    if (i >= B_SZ * HDIM) return;
    int b = i / HDIM, j = i % HDIM;
    const _Float16* p = h + (size_t)b * A_AG * HDIM + j;
    float acc = 0.f;
    #pragma unroll 5
    for (int a = 0; a < A_AG; ++a) acc += (float)p[(size_t)a * HDIM];
    c[i] = (_Float16)(acc * (1.0f / A_AG));
}

// ---------------------------------------------------------------------------
// Wave-tile WMMA GEMM:  out[M,N] = epilogue( A[M,KDIM] @ BT[N,KDIM]^T )
//   - A row-major f16 (global), BT = weight pre-transposed to [N][KDIM] f16,
//     cooperatively staged into LDS (dynamic smem, N*KDIM*2 bytes).
//   - one wave32 per 32x64 output tile; 8 WMMAs per 32-wide K step.
//   - epilogue: + bias[n], + addend[m/ADIV][n], fast_tanh, f16 or f32 store.
//
// Fragment layouts per CDNA5 ISA 7.12.2 (wave32):
//   A 16x32 f16 : lane L holds row M=L&15; K = (L>>4)*8 + {0..7, 16..23}
//   B 32x16 f16 : lane L holds col N=L&15; K = (L>>4)*8 + {0..7, 16..23}
//   C/D 16x16 f32: lane 0-15 -> N=lane, M=r; lane 16-31 -> N=lane-16, M=8+r.
template<int KDIM, bool OUT_F16, bool TANH, bool HAS_BIAS, bool HAS_ADD>
__global__ void gemm_wmma(const _Float16* __restrict__ Am,
                          const _Float16* __restrict__ BT,
                          const float* __restrict__ bias,
                          const float* __restrict__ addend,
                          int addend_div,
                          void* __restrict__ out,
                          int M, int N) {
    extern __shared__ _Float16 smemB[];           // [N][KDIM]

    // ---- cooperative stage of the full weight matrix into LDS ----
    {
        const int total_h = N * KDIM;             // halves; multiple of 8*blockDim
        for (int idx = threadIdx.x * 8; idx < total_h; idx += blockDim.x * 8)
            *(v8h*)(smemB + idx) = *(const v8h*)(BT + idx);
    }
    __syncthreads();

    const int wave    = (int)((blockIdx.x * blockDim.x + threadIdx.x) >> 5);
    const int lane    = threadIdx.x & 31;
    const int ngroups = N >> 6;                   // 64-wide column groups
    const int total   = (M >> 5) * ngroups;       // 32-row x 64-col wave tiles
    if (wave >= total) return;

    const int mgrp    = wave / ngroups;           // consecutive waves share M rows
    const int ngrp    = wave - mgrp * ngroups;
    const int l15     = lane & 15;
    const int halfsel = lane >> 4;

    // A: row fixed per lane per m-tile, kbase = halfsel*8 baked into the pointer.
    const _Float16* Arow0 = Am + (size_t)(mgrp * 32 + l15) * KDIM + halfsel * 8;
    const _Float16* Arow1 = Arow0 + (size_t)16 * KDIM;
    // B (LDS): col fixed per lane per n-tile, koff = halfsel*8 baked in.
    const _Float16* Brow = smemB + (ngrp * 64 + l15) * KDIM + halfsel * 8;

    v8f acc[MT][NT];
    #pragma unroll
    for (int mi = 0; mi < MT; ++mi)
        #pragma unroll
        for (int j = 0; j < NT; ++j) acc[mi][j] = (v8f){};

    // Unroll by 2 only: bounds the scheduler's pipelining window so the
    // 64 accumulator VGPRs + in-flight fragments stay within budget (no spills).
    #pragma unroll 2
    for (int kk = 0; kk < KDIM; kk += 32) {
        v16h a[MT];
        {
            v8h alo0 = *(const v8h*)(Arow0 + kk);       // K = kbase + 0..7
            v8h ahi0 = *(const v8h*)(Arow0 + kk + 16);  // K = kbase + 16..23
            v8h alo1 = *(const v8h*)(Arow1 + kk);
            v8h ahi1 = *(const v8h*)(Arow1 + kk + 16);
            #pragma unroll
            for (int i = 0; i < 8; ++i) {
                a[0][i] = alo0[i]; a[0][i + 8] = ahi0[i];
                a[1][i] = alo1[i]; a[1][i + 8] = ahi1[i];
            }
        }

        #pragma unroll
        for (int j = 0; j < NT; ++j) {
            const _Float16* bp = Brow + j * 16 * KDIM + kk;
            v8h blo = *(const v8h*)(bp);           // ds_load_b128
            v8h bhi = *(const v8h*)(bp + 16);      // ds_load_b128
            v16h b;
            #pragma unroll
            for (int i = 0; i < 8; ++i) { b[i] = blo[i]; b[i + 8] = bhi[i]; }
            #pragma unroll
            for (int mi = 0; mi < MT; ++mi)
                acc[mi][j] = __builtin_amdgcn_wmma_f32_16x16x32_f16(
                    false, a[mi], false, b, (short)0, acc[mi][j], false, false);
        }
    }

    const int rbase = halfsel * 8;
    #pragma unroll
    for (int j = 0; j < NT; ++j) {
        const int ncol = ngrp * 64 + j * 16 + l15;
        float bval = 0.f;
        if (HAS_BIAS) bval = bias[ncol];
        #pragma unroll
        for (int mi = 0; mi < MT; ++mi) {
            #pragma unroll
            for (int r = 0; r < 8; ++r) {
                const int m = mgrp * 32 + mi * 16 + rbase + r;
                float v = acc[mi][j][r] + bval;
                if (HAS_ADD) v += addend[(size_t)(m / addend_div) * N + ncol];
                if (TANH)    v = fast_tanh(v);
                if (OUT_F16) ((_Float16*)out)[(size_t)m * N + ncol] = (_Float16)v;
                else         ((float*)out)[(size_t)m * N + ncol]    = v;
            }
        }
    }
}

// ---------------------------------------------------------------------------
static inline int blocks_for_waves(int M, int N) {
    int waves = (M >> 5) * (N >> 6);      // one wave per 32x64 tile
    return (waves + 7) / 8;               // 8 waves per 256-thread block
}

extern "C" void kernel_launch(void* const* d_in, const int* in_sizes, int n_in,
                              void* d_out, int out_size, void* d_ws, size_t ws_size,
                              hipStream_t stream) {
    const float* x     = (const float*)d_in[0];  // [B,A,DIN]
    const float* W_enc = (const float*)d_in[1];  // [DIN,H]
    const float* b_enc = (const float*)d_in[2];  // [H]
    const float* W_h   = (const float*)d_in[3];  // [K,2H,H]
    const float* b_h   = (const float*)d_in[4];  // [K,H]
    const float* W_dec = (const float*)d_in[5];  // [H,DOUT]
    const float* b_dec = (const float*)d_in[6];  // [DOUT]
    float* out = (float*)d_out;                  // [B,A,DOUT]

    // ---- workspace layout (256B aligned) ----
    char* p = (char*)d_ws;
    auto take = [&](size_t bytes) {
        char* r = p;
        p += (bytes + 255) & ~(size_t)255;
        return r;
    };
    _Float16* xh      = (_Float16*)take((size_t)MROWS * DIN * 2);        // 52.4 MB
    _Float16* hbuf0   = (_Float16*)take((size_t)MROWS * HDIM * 2);       // 104.9 MB
    _Float16* hbuf1   = (_Float16*)take((size_t)MROWS * HDIM * 2);       // 104.9 MB
    _Float16* cmean   = (_Float16*)take((size_t)B_SZ * HDIM * 2);        // 2 MB
    float*    cpart   = (float*)   take((size_t)B_SZ * HDIM * 4);        // 4 MB
    _Float16* WencT   = (_Float16*)take((size_t)HDIM * DIN * 2);
    _Float16* WtopT   = (_Float16*)take((size_t)KCOMM * HDIM * HDIM * 2);
    _Float16* WbotT   = (_Float16*)take((size_t)KCOMM * HDIM * HDIM * 2);
    _Float16* WdecT   = (_Float16*)take((size_t)DOUT * HDIM * 2);
    (void)ws_size; (void)in_sizes; (void)n_in; (void)out_size;

    // ---- stage 0: convert x, transpose+convert all weights ----
    {
        long n = (long)MROWS * DIN;
        cvt_f32_to_f16<<<(int)((n + 255) / 256), 256, 0, stream>>>(x, xh, n);

        transpose_f32_to_f16<<<(DIN * HDIM + 255) / 256, 256, 0, stream>>>(
            W_enc, WencT, DIN, HDIM);
        for (int k = 0; k < KCOMM; ++k) {
            const float* wk = W_h + (size_t)k * 2 * HDIM * HDIM;
            transpose_f32_to_f16<<<(HDIM * HDIM + 255) / 256, 256, 0, stream>>>(
                wk, WtopT + (size_t)k * HDIM * HDIM, HDIM, HDIM);            // rows 0..H-1  (h part)
            transpose_f32_to_f16<<<(HDIM * HDIM + 255) / 256, 256, 0, stream>>>(
                wk + (size_t)HDIM * HDIM, WbotT + (size_t)k * HDIM * HDIM,   // rows H..2H-1 (c part)
                HDIM, HDIM);
        }
        transpose_f32_to_f16<<<(HDIM * DOUT + 255) / 256, 256, 0, stream>>>(
            W_dec, WdecT, HDIM, DOUT);
    }

    _Float16* hbufs[2] = {hbuf0, hbuf1};
    int cur = 0;

    // ---- encoder: h = tanh(xh @ WencT^T + b_enc), f16 out ----
    gemm_wmma<DIN, /*f16*/true, /*tanh*/true, /*bias*/true, /*add*/false>
        <<<blocks_for_waves(MROWS, HDIM), 256, HDIM * DIN * 2, stream>>>(
            xh, WencT, b_enc, nullptr, 1, hbufs[cur], MROWS, HDIM);

    // ---- comm steps ----
    for (int k = 0; k < KCOMM; ++k) {
        // c = mean over agents (per batch), f16
        mean_agents<<<(B_SZ * HDIM + 255) / 256, 256, 0, stream>>>(hbufs[cur], cmean);

        // cpart[b] = c[b] @ W_bot + b_h[k]   (4096x256 @ 256x256, f32 out, no tanh)
        gemm_wmma<HDIM, /*f16*/false, /*tanh*/false, /*bias*/true, /*add*/false>
            <<<blocks_for_waves(B_SZ, HDIM), 256, HDIM * HDIM * 2, stream>>>(
                cmean, WbotT + (size_t)k * HDIM * HDIM, b_h + (size_t)k * HDIM,
                nullptr, 1, cpart, B_SZ, HDIM);

        // h' = tanh(h @ W_top + cpart[row / A])   (204800x256 @ 256x256, f16 out)
        gemm_wmma<HDIM, /*f16*/true, /*tanh*/true, /*bias*/false, /*add*/true>
            <<<blocks_for_waves(MROWS, HDIM), 256, HDIM * HDIM * 2, stream>>>(
                hbufs[cur], WtopT + (size_t)k * HDIM * HDIM, nullptr,
                cpart, A_AG, hbufs[1 - cur], MROWS, HDIM);
        cur ^= 1;
    }

    // ---- decoder: out = h @ WdecT^T + b_dec, f32 out ----
    gemm_wmma<HDIM, /*f16*/false, /*tanh*/false, /*bias*/true, /*add*/false>
        <<<blocks_for_waves(MROWS, DOUT), 256, DOUT * HDIM * 2, stream>>>(
            hbufs[cur], WdecT, b_dec, nullptr, 1, out, MROWS, DOUT);
}
